// GoalRGCN_23699629539372
// MI455X (gfx1250) — matrix operand order repaired and checked by reading, not compiled, not run on hardware
//
#include <hip/hip_runtime.h>
#include <cstdint>
#include <cstddef>

// ---------------------------------------------------------------------------
// RGCN (3 layers, 16 relations) + actor/critic heads for MI455X (gfx1250).
// Strategy:
//   * counting-sort nodes by relation -> 16-node same-relation tiles
//   * per-tile 16x128 @ 128x128 GEMM via V_WMMA_F32_16X16X32_BF16
//     (divergence-free operand loads; 2 column tiles per wave to reuse A)
//   * edge scatter via global_atomic_add_f32 (msg/agg live in 192MB L2)
// ---------------------------------------------------------------------------

typedef __attribute__((ext_vector_type(16))) __bf16 v16bf;
typedef __attribute__((ext_vector_type(8)))  float  v8f;

struct B32x2 { uint4 a; uint4 b; };   // 32 bytes == 16 bf16 operands

static __device__ __forceinline__ unsigned short f2bf(float f) {
    unsigned u = __float_as_uint(f);
    u += 0x7FFFu + ((u >> 16) & 1u);          // round-to-nearest-even
    return (unsigned short)(u >> 16);
}

// ---- bucketing: meta[0..15]=counts, meta[16..31]=cursors, meta[32..48]=offsets
__global__ void k_zero_meta(int* __restrict__ meta) {
    if (threadIdx.x < 64) meta[threadIdx.x] = 0;
}

__global__ void k_count(const int* __restrict__ cls, int* __restrict__ meta, int n) {
    int i = blockIdx.x * blockDim.x + threadIdx.x;
    if (i < n) atomicAdd(&meta[cls[i]], 1);
}

__global__ void k_scan(int* __restrict__ meta) {
    if (threadIdx.x == 0) {
        int off = 0;
        meta[32] = 0;
        for (int r = 0; r < 16; ++r) { off += meta[r]; meta[33 + r] = off; }
    }
}

__global__ void k_fill(const int* __restrict__ cls, int* __restrict__ meta,
                       int* __restrict__ node_idx, int n) {
    int i = blockIdx.x * blockDim.x + threadIdx.x;
    if (i < n) {
        int c = cls[i];
        int pos = meta[32 + c] + atomicAdd(&meta[16 + c], 1);
        node_idx[pos] = i;
    }
}

// ---- weight prep: W[r][k][n] f32 -> Wt[r][n][k] bf16 (B-operand friendly)
__global__ void k_wprep(const float* __restrict__ W, unsigned short* __restrict__ Wt,
                        int total) {
    int i = blockIdx.x * blockDim.x + threadIdx.x;
    if (i < total) {
        int r   = i >> 14;          // /16384
        int rem = i & 16383;
        int nn  = rem >> 7;         // /128
        int kk  = rem & 127;
        Wt[i] = f2bf(W[(r << 14) + (kk << 7) + nn]);
    }
}

// ---- layer 0 transform: one-hot input => msg[n,:] = W0[c][c,:] * norm[n]
__global__ void k_msg0(const int* __restrict__ cls, const float* __restrict__ norm,
                       const float* __restrict__ W0, float* __restrict__ msg, int nNodes) {
    int t = blockIdx.x * blockDim.x + threadIdx.x;
    int n = t >> 5;
    if (n >= nNodes) return;
    int lane = t & 31;
    int c = cls[n];
    float nv = norm[n];
    float4 v = ((const float4*)(W0 + (size_t)c * 2176))[lane];   // 2176 = 16*128 + 128
    float4 o = make_float4(v.x * nv, v.y * nv, v.z * nv, v.w * nv);
    ((float4*)(msg + (size_t)n * 128))[lane] = o;
}

// ---- zero a float buffer (float4 granularity)
__global__ void k_zero_f4(float4* __restrict__ p, int n4) {
    int i = blockIdx.x * blockDim.x + threadIdx.x;
    if (i < n4) p[i] = make_float4(0.f, 0.f, 0.f, 0.f);
}

// ---- edge scatter: agg[dst] += msg[src]  (one wave32 per edge, float4/lane)
__global__ void k_scatter(const int* __restrict__ src, const int* __restrict__ dst,
                          const float* __restrict__ msg, float* __restrict__ agg, int nE) {
    int t = blockIdx.x * blockDim.x + threadIdx.x;
    int e = t >> 5;
    if (e >= nE) return;
    int lane = t & 31;
    int s = src[e], d = dst[e];
    float4 v = ((const float4*)(msg + (size_t)s * 128))[lane];
    float* a = agg + (size_t)d * 128 + lane * 4;
    atomicAdd(a + 0, v.x);
    atomicAdd(a + 1, v.y);
    atomicAdd(a + 2, v.z);
    atomicAdd(a + 3, v.w);
}

// ---- relu + f32->bf16 conversion (ushort4 granularity)
__global__ void k_relu_bf16(const float* __restrict__ agg, unsigned short* __restrict__ h,
                            int n4) {
    int i = blockIdx.x * blockDim.x + threadIdx.x;
    if (i >= n4) return;
    float4 v = ((const float4*)agg)[i];
    ushort4 o;
    o.x = f2bf(fmaxf(v.x, 0.f));
    o.y = f2bf(fmaxf(v.y, 0.f));
    o.z = f2bf(fmaxf(v.z, 0.f));
    o.w = f2bf(fmaxf(v.w, 0.f));
    ((ushort4*)h)[i] = o;
}

// ---- WMMA matmul: per relation bucket, tiles of 16 nodes.
// block = 128 threads = 4 waves; wave w owns column tiles 2w and 2w+1,
// so each A fetch feeds two WMMAs.  All operand loads are divergence-free:
// rows past the bucket end clamp to the last valid node (their D rows are
// computed but never stored), keeping EXEC all-1s through the WMMA chain.
// A: h_bf16[node,128] row-major. B: Wt[r][n][k] (k contiguous).
// D(16x16 f32) scaled by norm[node] and scattered to msg[node, col].
__global__ void k_rgcn_wmma(const unsigned short* __restrict__ h,
                            const unsigned short* __restrict__ Wt,
                            const float* __restrict__ norm,
                            const int* __restrict__ node_idx,
                            const int* __restrict__ meta,
                            float* __restrict__ msg) {
    const int r    = blockIdx.y;
    const int beg  = meta[32 + r];
    const int end  = meta[33 + r];
    const int tile0 = beg + (int)blockIdx.x * 16;
    if (tile0 >= end) return;

    const int wave = threadIdx.x >> 5;     // 0..3 : owns column tiles 2w, 2w+1
    const int lane = threadIdx.x & 31;
    const int mrow = lane & 15;
    const int hi   = lane >> 4;            // 0 or 1 (half-wave)

    // node backing A row M = lane&15 (same for both half-waves); clamp in-range
    int nidx = tile0 + mrow;
    if (nidx >= end) nidx = end - 1;
    const int node = node_idx[nidx];

    const int ncol0 = wave * 32 + mrow;        // B lane column, tile 2w
    const int ncol1 = ncol0 + 16;              // B lane column, tile 2w+1
    const unsigned short* Bb0 = Wt + (((size_t)r * 128 + ncol0) << 7);
    const unsigned short* Bb1 = Wt + (((size_t)r * 128 + ncol1) << 7);
    const unsigned short* Abase = h + ((size_t)node << 7) + hi * 8;

    v8f acc0 = {};
    v8f acc1 = {};
#pragma unroll
    for (int kb = 0; kb < 4; ++kb) {
        // ---- A operand: 16-bit A 16x32 layout.
        // lane holds K = kb*32 + hi*8 + {0..7}  and  kb*32 + 16 + hi*8 + {0..7}
        const uint4* pa = (const uint4*)(Abase + kb * 32);
        B32x2 ta;
        ta.a = pa[0];          // K = base .. base+7
        ta.b = pa[2];          // K = base+16 .. base+23
        v16bf av = __builtin_bit_cast(v16bf, ta);

        // ---- B operands: 16-bit B 32x16 layout.
        // lane holds N = ncol, K = kb*32 + hi*8 + {0..7} and kb*32 + 16 + hi*8 + {0..7}
        const uint4* pb0 = (const uint4*)(Bb0 + kb * 32 + hi * 8);
        B32x2 tb0; tb0.a = pb0[0]; tb0.b = pb0[2];
        v16bf bv0 = __builtin_bit_cast(v16bf, tb0);

        const uint4* pb1 = (const uint4*)(Bb1 + kb * 32 + hi * 8);
        B32x2 tb1; tb1.a = pb1[0]; tb1.b = pb1[2];
        v16bf bv1 = __builtin_bit_cast(v16bf, tb1);

        acc0 = __builtin_amdgcn_wmma_f32_16x16x32_bf16(
            false, av, false, bv0, (short)0, acc0, false, false);
        acc1 = __builtin_amdgcn_wmma_f32_16x16x32_bf16(
            false, av, false, bv1, (short)0, acc1, false, false);
    }

    // ---- C/D 16x16 f32 layout: lane N = lane&15; VGPR v -> M = v + hi*8
#pragma unroll
    for (int v = 0; v < 8; ++v) {
        int m  = v + hi * 8;
        int ni = tile0 + m;
        if (ni < end) {
            int nd = node_idx[ni];
            float nv = norm[nd];
            msg[((size_t)nd << 7) + ncol0] = acc0[v] * nv;
            msg[((size_t)nd << 7) + ncol1] = acc1[v] * nv;
        }
    }
}

// ---- softmax(root) + actor/critic heads. Single block, 512 threads.
__global__ void k_heads(const float* __restrict__ agg,
                        const float* __restrict__ a1w, const float* __restrict__ a1b,
                        const float* __restrict__ a2w, const float* __restrict__ a2b,
                        const float* __restrict__ c1w, const float* __restrict__ c1b,
                        const float* __restrict__ c2w, const float* __restrict__ c2b,
                        float* __restrict__ out) {
    __shared__ float root[128];
    __shared__ float ta[512];
    __shared__ float tc[512];
    __shared__ float red[16];
    const int tid = threadIdx.x;

    if (tid < 128) root[tid] = agg[tid];
    __syncthreads();
    if (tid == 0) {                        // softmax over 128 dims of node 0
        float mx = root[0];
        for (int i = 1; i < 128; ++i) mx = fmaxf(mx, root[i]);
        float s = 0.f;
        for (int i = 0; i < 128; ++i) { float e = expf(root[i] - mx); root[i] = e; s += e; }
        float inv = 1.f / s;
        for (int i = 0; i < 128; ++i) root[i] *= inv;
    }
    __syncthreads();

    float accA = 0.f, accC = 0.f;
    for (int i = 0; i < 128; ++i) {
        float rv = root[i];
        accA += rv * a1w[i * 512 + tid];
        accC += rv * c1w[i * 512 + tid];
    }
    ta[tid] = fmaxf(accA + a1b[tid], 0.f);
    tc[tid] = fmaxf(accC + c1b[tid], 0.f);
    __syncthreads();

    if (tid < 128) {
        float p = 0.f;
        for (int j = 0; j < 512; ++j) p += ta[j] * a2w[j * 128 + tid];
        out[tid] = p + a2b[tid];
    }

    float v = tc[tid] * c2w[tid];
    for (int o = 16; o > 0; o >>= 1) v += __shfl_xor(v, o, 32);
    if ((tid & 31) == 0) red[tid >> 5] = v;
    __syncthreads();
    if (tid == 0) {
        float s = 0.f;
        for (int w = 0; w < 16; ++w) s += red[w];
        out[128] = s + c2b[0];
    }
}

// ---------------------------------------------------------------------------
extern "C" void kernel_launch(void* const* d_in, const int* in_sizes, int n_in,
                              void* d_out, int out_size, void* d_ws, size_t ws_size,
                              hipStream_t stream) {
    const int*   cls  = (const int*)d_in[0];
    const float* norm = (const float*)d_in[1];
    const int*   src  = (const int*)d_in[2];
    const int*   dst  = (const int*)d_in[3];
    const float* W0   = (const float*)d_in[4];
    const float* W1   = (const float*)d_in[5];
    const float* W2   = (const float*)d_in[6];
    const float* a1w  = (const float*)d_in[7];
    const float* a1b  = (const float*)d_in[8];
    const float* a2w  = (const float*)d_in[9];
    const float* a2b  = (const float*)d_in[10];
    const float* c1w  = (const float*)d_in[11];
    const float* c1b  = (const float*)d_in[12];
    const float* c2w  = (const float*)d_in[13];
    const float* c2b  = (const float*)d_in[14];
    float* out = (float*)d_out;

    const int N = in_sizes[0];
    const int E = in_sizes[2];

    // ---- carve workspace (256B aligned chunks)
    size_t off = 0;
    auto carve = [&](size_t bytes) -> char* {
        char* p = (char*)d_ws + off;
        off = (off + bytes + 255) & ~(size_t)255;
        return p;
    };
    float*          agg      = (float*)carve((size_t)N * 128 * sizeof(float));
    float*          msg      = (float*)carve((size_t)N * 128 * sizeof(float));
    unsigned short* hbf      = (unsigned short*)carve((size_t)N * 128 * sizeof(unsigned short));
    unsigned short* W1t      = (unsigned short*)carve((size_t)16 * 128 * 128 * sizeof(unsigned short));
    unsigned short* W2t      = (unsigned short*)carve((size_t)16 * 128 * 128 * sizeof(unsigned short));
    int*            node_idx = (int*)carve((size_t)N * sizeof(int));
    int*            meta     = (int*)carve(64 * sizeof(int));
    (void)ws_size; (void)n_in; (void)out_size;

    const int T = 256;
    const int nB_nodes = (N + T - 1) / T;
    const int n4       = N * 32;                  // N*128/4
    const int nB_feat  = (n4 + T - 1) / T;        // float4-granularity over [N,128]
    const int nB_edge  = (E * 32 + T - 1) / T;    // wave per edge
    const int wtotal   = 16 * 128 * 128;
    const int nB_w     = (wtotal + T - 1) / T;
    const dim3 mmGrid((unsigned)((N + 15) / 16), 16, 1);

    // ---- bucket nodes by relation
    k_zero_meta<<<1, 64, 0, stream>>>(meta);
    k_count<<<nB_nodes, T, 0, stream>>>(cls, meta, N);
    k_scan<<<1, 32, 0, stream>>>(meta);
    k_fill<<<nB_nodes, T, 0, stream>>>(cls, meta, node_idx, N);

    // ---- weight prep (f32 -> bf16 transposed)
    k_wprep<<<nB_w, T, 0, stream>>>(W1, W1t, wtotal);
    k_wprep<<<nB_w, T, 0, stream>>>(W2, W2t, wtotal);

    // ---- layer 0: gather + scatter + relu
    k_msg0<<<(N * 32 + T - 1) / T, T, 0, stream>>>(cls, norm, W0, msg, N);
    k_zero_f4<<<nB_feat, T, 0, stream>>>((float4*)agg, n4);
    k_scatter<<<nB_edge, T, 0, stream>>>(src, dst, msg, agg, E);
    k_relu_bf16<<<nB_feat, T, 0, stream>>>(agg, hbf, n4);

    // ---- layer 1: WMMA transform + scatter + relu
    k_rgcn_wmma<<<mmGrid, 128, 0, stream>>>(hbf, W1t, norm, node_idx, meta, msg);
    k_zero_f4<<<nB_feat, T, 0, stream>>>((float4*)agg, n4);
    k_scatter<<<nB_edge, T, 0, stream>>>(src, dst, msg, agg, E);
    k_relu_bf16<<<nB_feat, T, 0, stream>>>(agg, hbf, n4);

    // ---- layer 2: WMMA transform + scatter (softmax folded into heads)
    k_rgcn_wmma<<<mmGrid, 128, 0, stream>>>(hbf, W2t, norm, node_idx, meta, msg);
    k_zero_f4<<<nB_feat, T, 0, stream>>>((float4*)agg, n4);
    k_scatter<<<nB_edge, T, 0, stream>>>(src, dst, msg, agg, E);

    // ---- heads on root node (softmax + two MLPs)
    k_heads<<<1, 512, 0, stream>>>(agg, a1w, a1b, a2w, a2b, c1w, c1b, c2w, c2b, out);
}
